// RelationNetwork_43001212567988
// MI455X (gfx1250) — matrix operand order repaired and checked by reading, not compiled, not run on hardware
//
#include <hip/hip_runtime.h>

// ---------------- CDNA5 WMMA types ----------------
typedef __attribute__((ext_vector_type(16))) __bf16 v16bf;
typedef __attribute__((ext_vector_type(8)))  float  v8f;

#define HH 512          // hidden width (Hg = Og = Hf)
#define NN 40           // facts
#define BB 32           // batch
#define MT 64           // pair rows per workgroup tile
#define STR 520         // padded LDS row stride (bf16): 1040B -> +4 banks/row
#define NTILES 25       // 1600 / 64
#define XSTR 264        // padded LDS stride for x tile (528B -> +4 banks/row)

__device__ __forceinline__ unsigned short f2bf(float f) {
  unsigned int u = __float_as_uint(f);
  u += 0x7FFFu + ((u >> 16) & 1u);        // round-to-nearest-even
  return (unsigned short)(u >> 16);
}

__device__ __forceinline__ v16bf ld_frag(const unsigned short* p0, const unsigned short* p1) {
  v16bf v;
  uint4* q = reinterpret_cast<uint4*>(&v);
  q[0] = *reinterpret_cast<const uint4*>(p0);
  q[1] = *reinterpret_cast<const uint4*>(p1);
  return v;
}

// ---------------- prep: zero pooled embedding ----------------
__global__ void k_zero(float* p, int n) {
  int i = blockIdx.x * 256 + threadIdx.x;
  if (i < n) p[i] = 0.0f;
}

// ---- prep: W1,W2 -> bf16 [out=512][in=512]; W0ab -> bf16 [out=1024][in=256] ----
__global__ void k_wt(const float* __restrict__ w0, const float* __restrict__ w1,
                     const float* __restrict__ w2,
                     unsigned short* __restrict__ w0t, unsigned short* __restrict__ w1t,
                     unsigned short* __restrict__ w2t) {
  int i = blockIdx.x * 256 + threadIdx.x;      // 3 * 262144 elements
  int which = i >> 18;
  int j = i & 0x3FFFF;
  if (which == 0) {
    int k = j & 511, n = j >> 9;
    w1t[(size_t)n * HH + k] = f2bf(w1[(size_t)k * HH + n]);
  } else if (which == 1) {
    int k = j & 511, n = j >> 9;
    w2t[(size_t)n * HH + k] = f2bf(w2[(size_t)k * HH + n]);
  } else {
    int k = j & 255, n = j >> 8;               // n < 1024: [W0a | W0b] columns
    float v = (n < 512) ? w0[(size_t)k * HH + n]
                        : w0[(size_t)(256 + k) * HH + (n - 512)];
    w0t[(size_t)n * 256 + k] = f2bf(v);
  }
}

// ---------------- prep: [A | Bm] = x(1280x256) @ W0ab(256x1024), WMMA ----------------
__global__ __launch_bounds__(256) void k_ab(const float* __restrict__ x,
                                            const __bf16* __restrict__ W0t,
                                            float* __restrict__ A,
                                            float* __restrict__ Bm) {
  __shared__ unsigned short xt[MT * XSTR];     // 64 x 256 bf16, padded
  int t = threadIdx.x;
  int row0 = blockIdx.x * MT;                  // 20 blocks * 64 rows = 1280
  { // stage x tile as bf16
    int m = t >> 2;
    int k0 = (t & 3) * 64;
    const float* xr = x + ((size_t)(row0 + m)) * 256 + k0;
    unsigned int* dst = reinterpret_cast<unsigned int*>(&xt[(size_t)m * XSTR + k0]);
    #pragma unroll 4
    for (int k = 0; k < 64; k += 4) {
      float4 v = *reinterpret_cast<const float4*>(xr + k);
      dst[(k >> 1) + 0] = (unsigned)f2bf(v.x) | ((unsigned)f2bf(v.y) << 16);
      dst[(k >> 1) + 1] = (unsigned)f2bf(v.z) | ((unsigned)f2bf(v.w) << 16);
    }
  }
  __syncthreads();

  int lane = t & 31, wave = t >> 5;
  int ncol = lane & 15, khalf = lane >> 4;

  #pragma unroll 1
  for (int nt = 0; nt < 8; ++nt) {             // 64 N-tiles, 8 per wave
    int n0 = (nt * 8 + wave) * 16;
    v8f acc[4] = {{}, {}, {}, {}};
    const __bf16* bcol = W0t + (size_t)(n0 + ncol) * 256 + khalf * 16;
    #pragma unroll 4
    for (int ks = 0; ks < 8; ++ks) {           // K = 256
      v16bf bv = ld_frag((const unsigned short*)(bcol + ks * 32),
                         (const unsigned short*)(bcol + ks * 32 + 8));
      const unsigned short* ab = xt + khalf * 16 + ks * 32;
      #pragma unroll
      for (int mb = 0; mb < 4; ++mb) {
        const unsigned short* ar = ab + (size_t)(mb * 16 + ncol) * XSTR;
        v16bf av = ld_frag(ar, ar + 8);
        acc[mb] = __builtin_amdgcn_wmma_f32_16x16x32_bf16(
            false, av, false, bv, (short)0, acc[mb], false, false);
      }
    }
    float* outp = (n0 < 512) ? A : Bm;
    int n = (n0 < 512) ? (n0 + ncol) : (n0 - 512 + ncol);
    #pragma unroll
    for (int mb = 0; mb < 4; ++mb) {
      int mrow = row0 + mb * 16 + khalf * 8;
      #pragma unroll
      for (int r = 0; r < 8; ++r)
        outp[(size_t)(mrow + r) * HH + n] = acc[mb][r];
    }
  }
}

// ---------------- prep: C = q@W0[512:768] + b0 ----------------
__global__ void k_c(const float* __restrict__ q, const float* __restrict__ g_w0,
                    const float* __restrict__ g_b0, float* __restrict__ C) {
  int i = blockIdx.x * 256 + threadIdx.x;      // 32*512
  int b = i >> 9, h = i & 511;
  const float* W = g_w0 + (size_t)512 * HH;
  float s = g_b0[h];
  #pragma unroll 4
  for (int d = 0; d < 256; ++d) s += q[(size_t)b * 256 + d] * W[(size_t)d * HH + h];
  C[i] = s;
}

// ---------------- fused g-MLP WMMA layer ----------------
// Each wave owns ALL 4 M-blocks and 4 N-tiles: one B-fragment load feeds 4 WMMAs,
// cutting weight L2 traffic 4x vs an M-split. A-fragments reload from LDS (cheap).
__device__ __forceinline__ void gemm_layer(const unsigned short* __restrict__ src,
                                           const __bf16* __restrict__ W,
                                           const float* __restrict__ bias,
                                           unsigned short* __restrict__ dst,
                                           float* __restrict__ embRow,
                                           int wave, int ncol, int khalf) {
  #pragma unroll 1
  for (int nt = 0; nt < 4; ++nt) {             // 32 N-tiles, 4 per wave
    int n0 = (nt * 8 + wave) * 16;
    v8f acc[4] = {{}, {}, {}, {}};
    const __bf16* bcol = W + (size_t)(n0 + ncol) * HH + khalf * 16;
    #pragma unroll 4
    for (int ks = 0; ks < 16; ++ks) {          // K = 512
      v16bf bv = ld_frag((const unsigned short*)(bcol + ks * 32),
                         (const unsigned short*)(bcol + ks * 32 + 8));
      const unsigned short* ab = src + khalf * 16 + ks * 32;
      #pragma unroll
      for (int mb = 0; mb < 4; ++mb) {
        const unsigned short* ar = ab + (size_t)(mb * 16 + ncol) * STR;
        v16bf av = ld_frag(ar, ar + 8);
        acc[mb] = __builtin_amdgcn_wmma_f32_16x16x32_bf16(
            false, av, false, bv, (short)0, acc[mb], false, false);
      }
    }
    float bs = bias[n0 + ncol];
    if (dst) {
      #pragma unroll
      for (int mb = 0; mb < 4; ++mb) {
        unsigned short* d = dst + (size_t)(mb * 16 + khalf * 8) * STR + n0 + ncol;
        #pragma unroll
        for (int r = 0; r < 8; ++r) d[(size_t)r * STR] = f2bf(fmaxf(acc[mb][r] + bs, 0.0f));
      }
    } else {
      float s = 0.0f;
      #pragma unroll
      for (int mb = 0; mb < 4; ++mb)
        #pragma unroll
        for (int r = 0; r < 8; ++r) s += fmaxf(acc[mb][r] + bs, 0.0f);
      atomicAdd(embRow + n0 + ncol, s);        // lanes L and L+16: disjoint row halves
    }
  }
}

// ---------------- main fused kernel ----------------
__global__ __launch_bounds__(256) void rn_main(const float* __restrict__ A,
                                               const float* __restrict__ Bm,
                                               const float* __restrict__ C,
                                               const __bf16* __restrict__ W1t,
                                               const __bf16* __restrict__ W2t,
                                               const float* __restrict__ b1,
                                               const float* __restrict__ b2,
                                               float* __restrict__ emb) {
  extern __shared__ unsigned short smem[];     // 2 * MT*STR bf16 = 133120 B
  unsigned short* h0 = smem;
  unsigned short* h1 = smem + MT * STR;

  int b  = blockIdx.x / NTILES;
  int p0 = (blockIdx.x % NTILES) * MT;
  int t  = threadIdx.x;

  // --- h0[m][k] = relu(A[b, p%N] + Bm[b, p/N] + C[b]) as bf16 ---
  {
    int m  = t >> 2;
    int k0 = (t & 3) * 128;
    int p  = p0 + m;
    int i  = p % NN, j = p / NN;
    const float* Ar = A + ((size_t)b * NN + i) * HH + k0;
    const float* Br = Bm + ((size_t)b * NN + j) * HH + k0;
    const float* Cr = C + (size_t)b * HH + k0;
    unsigned int* dst = reinterpret_cast<unsigned int*>(&h0[(size_t)m * STR + k0]);
    #pragma unroll 4
    for (int k = 0; k < 128; k += 4) {
      float4 a4 = *reinterpret_cast<const float4*>(Ar + k);
      float4 b4 = *reinterpret_cast<const float4*>(Br + k);
      float4 c4 = *reinterpret_cast<const float4*>(Cr + k);
      float f0 = fmaxf(a4.x + b4.x + c4.x, 0.0f);
      float f1 = fmaxf(a4.y + b4.y + c4.y, 0.0f);
      float f2 = fmaxf(a4.z + b4.z + c4.z, 0.0f);
      float f3 = fmaxf(a4.w + b4.w + c4.w, 0.0f);
      dst[(k >> 1) + 0] = (unsigned)f2bf(f0) | ((unsigned)f2bf(f1) << 16);
      dst[(k >> 1) + 1] = (unsigned)f2bf(f2) | ((unsigned)f2bf(f3) << 16);
    }
  }
  __syncthreads();

  int lane = t & 31, wave = t >> 5;
  int ncol = lane & 15, khalf = lane >> 4;

  // layer g1: h0 -> h1
  gemm_layer(h0, W1t, b1, h1, nullptr, wave, ncol, khalf);
  __syncthreads();
  // layer g2: h1 -> sum-pool into emb[b]
  gemm_layer(h1, W2t, b2, nullptr, emb + (size_t)b * HH, wave, ncol, khalf);
}

// ---------------- f-MLP: emb -> out ----------------
__global__ __launch_bounds__(256) void k_f(const float* __restrict__ emb,
                                           const float* __restrict__ f_w0,
                                           const float* __restrict__ f_b0,
                                           const float* __restrict__ f_w1,
                                           const float* __restrict__ f_b1,
                                           float* __restrict__ out) {
  __shared__ float hf[HH];
  int b = blockIdx.x;
  for (int h = threadIdx.x; h < HH; h += 256) {
    float s = f_b0[h];
    #pragma unroll 4
    for (int in = 0; in < HH; ++in) s += emb[(size_t)b * HH + in] * f_w0[(size_t)in * HH + h];
    hf[h] = fmaxf(s, 0.0f);
  }
  __syncthreads();
  int o = threadIdx.x;   // 256 outputs
  float s = f_b1[o];
  #pragma unroll 4
  for (int in = 0; in < HH; ++in) s += hf[in] * f_w1[(size_t)in * 256 + o];
  out[(size_t)b * 256 + o] = fmaxf(s, 0.0f);
}

// ---------------- launch ----------------
extern "C" void kernel_launch(void* const* d_in, const int* in_sizes, int n_in,
                              void* d_out, int out_size, void* d_ws, size_t ws_size,
                              hipStream_t stream) {
  const float* x    = (const float*)d_in[0];
  const float* q    = (const float*)d_in[1];
  const float* g_w0 = (const float*)d_in[2];
  const float* g_b0 = (const float*)d_in[3];
  const float* g_w1 = (const float*)d_in[4];
  const float* g_b1 = (const float*)d_in[5];
  const float* g_w2 = (const float*)d_in[6];
  const float* g_b2 = (const float*)d_in[7];
  const float* f_w0 = (const float*)d_in[8];
  const float* f_b0 = (const float*)d_in[9];
  const float* f_w1 = (const float*)d_in[10];
  const float* f_b1 = (const float*)d_in[11];

  char* ws = (char*)d_ws;
  float*          A   = (float*)(ws + 0);                 // 1280*512 f32
  float*          Bm  = (float*)(ws + 2621440);           // 1280*512 f32
  float*          C   = (float*)(ws + 5242880);           // 32*512 f32
  float*          emb = (float*)(ws + 5308416);           // 32*512 f32
  unsigned short* W1t = (unsigned short*)(ws + 5373952);  // 512*512 bf16
  unsigned short* W2t = (unsigned short*)(ws + 5898240);  // 512*512 bf16
  unsigned short* W0t = (unsigned short*)(ws + 6422528);  // 1024*256 bf16

  k_zero<<<64, 256, 0, stream>>>(emb, BB * HH);
  k_wt  <<<3072, 256, 0, stream>>>(g_w0, g_w1, g_w2, W0t, W1t, W2t);
  k_ab  <<<20, 256, 0, stream>>>(x, (const __bf16*)W0t, A, Bm);
  k_c   <<<64, 256, 0, stream>>>(q, g_w0, g_b0, C);

  size_t lds = (size_t)2 * MT * STR * sizeof(unsigned short);  // 133120 B
  rn_main<<<BB * NTILES, 256, lds, stream>>>(A, Bm, C,
                                             (const __bf16*)W1t, (const __bf16*)W2t,
                                             g_b1, g_b2, emb);

  k_f<<<BB, 256, 0, stream>>>(emb, f_w0, f_b0, f_w1, f_b1, (float*)d_out);
}
